// ScaledDotProductWithBoxAttention_65566970741295
// MI455X (gfx1250) — matrix operand order repaired and checked by reading, not compiled, not run on hardware
//
#include <hip/hip_runtime.h>

// ---------------------------------------------------------------------------
// CDNA5 (gfx1250, wave32) WMMA implementation of box-attention.
//   B=512, N=49, DM=1024, H=16, DK=DV=64, Mtot = B*N = 25088
// Pipeline:
//   1) gemm_bias<f32 in, f16 out>  x3 : Q/K/V projections (25088x1024x1024)
//   2) box_attention kernel            : one workgroup per (b,h), all matrix
//                                        products via v_wmma_f32_16x16x32_f16
//   3) gemm_bias<f16 in, f32 out>      : output projection -> d_out
// Workspace: 4 * 25088*1024 f16 tensors = ~196 MB.
// ---------------------------------------------------------------------------

typedef __attribute__((ext_vector_type(16))) _Float16 v16h;
typedef __attribute__((ext_vector_type(8)))  _Float16 h8v;
typedef __attribute__((ext_vector_type(4)))  _Float16 h4v;
typedef __attribute__((ext_vector_type(2)))  _Float16 h2v;
typedef __attribute__((ext_vector_type(2)))  __fp16   pk2v;  // cvt_pkrtz native type
typedef __attribute__((ext_vector_type(4)))  __fp16   pk4v;
typedef __attribute__((ext_vector_type(8)))  float    v8f;

#define WMMA_F16(acc, a, b)                                                  \
  (acc) = __builtin_amdgcn_wmma_f32_16x16x32_f16(false, (a), false, (b),     \
                                                 (short)0, (acc), false, false)

// A fragment (16x32 f16, row-major LDS tile [row][k], stride in halves).
__device__ __forceinline__ v16h load_a_frag(const _Float16* tile, int stride, int lane) {
  const _Float16* p = tile + (size_t)(lane & 15) * stride + ((lane >> 4) << 3);
  h8v lo = *(const h8v*)(p);
  h8v hi = *(const h8v*)(p + 16);
  return __builtin_shufflevector(lo, hi, 0,1,2,3,4,5,6,7,8,9,10,11,12,13,14,15);
}

// B fragment (32x16 f16). LDS tile stored transposed: [n][k], stride in halves.
__device__ __forceinline__ v16h load_b_frag(const _Float16* tile, int stride, int lane) {
  const _Float16* p = tile + (size_t)(lane & 15) * stride + ((lane >> 4) << 3);
  h8v lo = *(const h8v*)(p);
  h8v hi = *(const h8v*)(p + 16);
  return __builtin_shufflevector(lo, hi, 0,1,2,3,4,5,6,7,8,9,10,11,12,13,14,15);
}

__device__ __forceinline__ v8f zero8() {
  v8f z = {0.f, 0.f, 0.f, 0.f, 0.f, 0.f, 0.f, 0.f};
  return z;
}

// packed f32->f16 conversion: two v_cvt_pk_rtz_f16_f32, bit-cast to _Float16x4
__device__ __forceinline__ h4v pack4(float x, float y, float z, float w) {
  pk2v lo = __builtin_amdgcn_cvt_pkrtz(x, y);
  pk2v hi = __builtin_amdgcn_cvt_pkrtz(z, w);
  pk4v v = __builtin_shufflevector(lo, hi, 0, 1, 2, 3);
  return __builtin_bit_cast(h4v, v);
}

// ---------------------------------------------------------------------------
// GEMM: C[M,P] = A[M,K] @ W[K,P] + bias.  Block tile 128x128, 8 waves of 32x64.
// Software pipelined: next K-tile prefetched into registers during WMMA.
// ---------------------------------------------------------------------------
template <typename AT, typename CT>
__global__ __launch_bounds__(256)
void gemm_bias_kernel(const AT* __restrict__ A, const float* __restrict__ W,
                      const float* __restrict__ bias, CT* __restrict__ C,
                      int Kdim, int Pdim) {
  constexpr int BM = 128, BP = 128, KS = 32, STR = KS + 8;  // 40 halves = 80B
  __shared__ __align__(16) _Float16 As[BM * STR];
  __shared__ __align__(16) _Float16 Bs[BP * STR];  // transposed W tile: [p][k]

  const int tid  = threadIdx.x;
  const int lane = tid & 31;
  const int wid  = tid >> 5;
  const int wm   = wid >> 1;  // 0..3
  const int wn   = wid & 1;   // 0..1
  const size_t m0 = (size_t)blockIdx.y * BM;
  const int    p0 = blockIdx.x * BP;

  // staging register buffers (one K-tile in flight)
  float4 aF[4];  // A path, f32 source
  h8v    aH[2];  // A path, f16 source
  float4 wF[4];

  const int arF = tid >> 3, acF = (tid & 7) * 4;  // f32 A mapping (32 rows/pass)
  const int arH = tid >> 2, acH = (tid & 3) * 8;  // f16 A mapping (64 rows/pass)
  const int wkr = tid >> 5, wc  = (tid & 31) * 4; // W mapping (8 k-rows/pass)

  auto load_tiles = [&](int k0) {
    if (sizeof(AT) == 4) {
#pragma unroll
      for (int i = 0; i < 4; i++)
        aF[i] = *(const float4*)((const float*)A + (m0 + arF + 32 * i) * Kdim + k0 + acF);
    } else {
#pragma unroll
      for (int i = 0; i < 2; i++)
        aH[i] = *(const h8v*)((const _Float16*)A + (m0 + arH + 64 * i) * Kdim + k0 + acH);
    }
#pragma unroll
    for (int i = 0; i < 4; i++)
      wF[i] = *(const float4*)(W + (size_t)(k0 + wkr + 8 * i) * Pdim + p0 + wc);
  };

  auto store_tiles = [&]() {
    if (sizeof(AT) == 4) {
#pragma unroll
      for (int i = 0; i < 4; i++)
        *(h4v*)&As[(arF + 32 * i) * STR + acF] = pack4(aF[i].x, aF[i].y, aF[i].z, aF[i].w);
    } else {
#pragma unroll
      for (int i = 0; i < 2; i++)
        *(h8v*)&As[(arH + 64 * i) * STR + acH] = aH[i];
    }
#pragma unroll
    for (int i = 0; i < 4; i++) {
      const pk2v lo = __builtin_amdgcn_cvt_pkrtz(wF[i].x, wF[i].y);
      const pk2v hi = __builtin_amdgcn_cvt_pkrtz(wF[i].z, wF[i].w);
      const int kr = wkr + 8 * i;
      Bs[(wc + 0) * STR + kr] = (_Float16)(float)lo[0];
      Bs[(wc + 1) * STR + kr] = (_Float16)(float)lo[1];
      Bs[(wc + 2) * STR + kr] = (_Float16)(float)hi[0];
      Bs[(wc + 3) * STR + kr] = (_Float16)(float)hi[1];
    }
  };

  v8f acc[2][4];
#pragma unroll
  for (int i = 0; i < 2; i++)
#pragma unroll
    for (int j = 0; j < 4; j++) acc[i][j] = zero8();

  // prologue: stage first K-tile
  load_tiles(0);
  store_tiles();
  __syncthreads();

  for (int k0 = 0; k0 < Kdim; k0 += KS) {
    const bool hasNext = (k0 + KS) < Kdim;
    if (hasNext) {
      load_tiles(k0 + KS);  // global loads in flight while WMMA runs below
      if (k0 + 2 * KS < Kdim) {  // distance-2 prefetch into GL2
        __builtin_prefetch((const void*)((const AT*)A + (m0 + arF) * Kdim + k0 + 2 * KS), 0, 1);
        __builtin_prefetch((const void*)(W + (size_t)(k0 + 2 * KS + wkr) * Pdim + p0 + wc), 0, 1);
      }
    }

    v16h afrag[2], bfrag[4];
#pragma unroll
    for (int i = 0; i < 2; i++)
      afrag[i] = load_a_frag(&As[(wm * 32 + i * 16) * STR], STR, lane);
#pragma unroll
    for (int j = 0; j < 4; j++)
      bfrag[j] = load_b_frag(&Bs[(wn * 64 + j * 16) * STR], STR, lane);
#pragma unroll
    for (int i = 0; i < 2; i++)
#pragma unroll
      for (int j = 0; j < 4; j++) WMMA_F16(acc[i][j], afrag[i], bfrag[j]);

    if (hasNext) {
      __syncthreads();   // all waves done reading current LDS tile
      store_tiles();     // commit prefetched registers
      __syncthreads();   // LDS tile ready
    }
  }

  // epilogue: bias + store (C/D layout: vgpr e -> M = e + 8*(lane>=16))
#pragma unroll
  for (int i = 0; i < 2; i++) {
    const int rbase = wm * 32 + i * 16 + ((lane >> 4) << 3);
#pragma unroll
    for (int j = 0; j < 4; j++) {
      const int col = p0 + wn * 64 + j * 16 + (lane & 15);
      const float bv = bias[col];
#pragma unroll
      for (int e = 0; e < 8; e++) {
        const size_t row = m0 + rbase + e;
        C[row * Pdim + col] = (CT)(acc[i][j][e] + bv);
      }
    }
  }
}

// ---------------------------------------------------------------------------
// Box attention: one workgroup (256 thr / 8 waves) per (b,h). N=49 padded to 64.
// ---------------------------------------------------------------------------
__global__ __launch_bounds__(256)
void box_attention_kernel(const _Float16* __restrict__ Qg,
                          const _Float16* __restrict__ Kg,
                          const _Float16* __restrict__ Vg,
                          const float* __restrict__ posg,
                          const float* __restrict__ boxg,
                          const float* __restrict__ Wattg,
                          const float* __restrict__ battg,
                          const float* __restrict__ spg,
                          _Float16* __restrict__ AOg) {
  constexpr int Nn = 49, S = 72, SF = 68;  // S: f16 stride (144B, 16B-aligned)
  const int h = blockIdx.x, b = blockIdx.y;
  const int tid = threadIdx.x, lane = tid & 31, wid = tid >> 5;

  __shared__ __align__(16) _Float16 Qs[64 * S];   // [q][d]
  __shared__ __align__(16) _Float16 Ks[64 * S];   // [kk][d]
  __shared__ __align__(16) _Float16 Ps[64 * S];   // [n][d]
  __shared__ __align__(16) _Float16 Vt[64 * S];   // [dv][kk] (V transposed)
  __shared__ __align__(16) _Float16 G4s[64 * S];  // P@P^T f16, A-operand for @Watt
  __shared__ __align__(16) _Float16 WtT[64 * S];  // [n][k] = Watt[k][n]
  __shared__ __align__(16) _Float16 S16[64 * S];  // softmaxed probs f16
  __shared__ float Sf[64 * SF];
  __shared__ float qn[64], kn[64], battS[64];     // qn/kn hold RECIPROCAL norms
  __shared__ float scaleS;

  const size_t gbase = (size_t)(b * Nn) * 1024 + (size_t)h * 64;

  // ---------------- load phase: fill padded f16 tiles (zeros outside 49) ----
  {
    const int r = tid >> 2;            // 0..63
    const int c16 = (tid & 3) << 4;    // 0,16,32,48
    const bool rv = (r < Nn);
    const size_t rowoff = gbase + (size_t)r * 1024 + c16;
    const _Float16 z0 = (_Float16)0.f;
    h8v zh = {z0, z0, z0, z0, z0, z0, z0, z0};
    h8v q0 = zh, q1 = zh, k0v = zh, k1v = zh, v0 = zh, v1 = zh;
    if (rv) {
      q0 = *(const h8v*)(Qg + rowoff);  q1 = *(const h8v*)(Qg + rowoff + 8);
      k0v = *(const h8v*)(Kg + rowoff); k1v = *(const h8v*)(Kg + rowoff + 8);
      v0 = *(const h8v*)(Vg + rowoff);  v1 = *(const h8v*)(Vg + rowoff + 8);
    }
    *(h8v*)&Qs[r * S + c16] = q0;  *(h8v*)&Qs[r * S + c16 + 8] = q1;
    *(h8v*)&Ks[r * S + c16] = k0v; *(h8v*)&Ks[r * S + c16 + 8] = k1v;
#pragma unroll
    for (int j = 0; j < 8; j++) {           // V transposed: [dv][kk]
      Vt[(c16 + j) * S + r] = v0[j];
      Vt[(c16 + 8 + j) * S + r] = v1[j];
    }
#pragma unroll
    for (int j = 0; j < 4; j++) {           // grids_pos f32 -> f16 (packed cvt)
      float4 pv = {0.f, 0.f, 0.f, 0.f};
      if (rv) pv = *(const float4*)(posg + rowoff + j * 4);
      *(h4v*)&Ps[r * S + c16 + j * 4] = pack4(pv.x, pv.y, pv.z, pv.w);
    }
#pragma unroll
    for (int j = 0; j < 16; j++) {          // Watt transposed: WtT[n][k]
      const int cc = c16 + j;
      const float w = (rv && cc < Nn) ? Wattg[r * Nn + cc] : 0.f;
      WtT[cc * S + r] = (_Float16)w;
    }
    if (tid < 64) battS[tid] = (tid < Nn) ? battg[tid] : 0.f;
    if (tid == 0) scaleS = __expf(fminf(spg[h], 4.6051702f));  // log(100)
  }
  __syncthreads();

  // ---------------- reciprocal row norms of Q and K (cosine term) ----------
  if (tid < 64) {
    float sq = 0.f, sk = 0.f;
    for (int d = 0; d < 64; d++) {
      const float a = (float)Qs[tid * S + d];
      const float c = (float)Ks[tid * S + d];
      sq += a * a; sk += c * c;
    }
    qn[tid] = __builtin_amdgcn_rcpf(fmaxf(sqrtf(sq), 1e-12f));
    kn[tid] = __builtin_amdgcn_rcpf(fmaxf(sqrtf(sk), 1e-12f));
  }
  __syncthreads();

  // ---------------- stage 1: G1+G2, Q@K^T, P@P^T (WMMA, 2 tiles/wave) ------
#pragma unroll
  for (int t = 0; t < 2; t++) {
    const int tile = wid * 2 + t;            // 0..15
    const int tm = tile >> 2, tn = tile & 3;
    v8f acc12 = zero8(), accQK = zero8(), accPP = zero8();
#pragma unroll
    for (int kk = 0; kk < 2; kk++) {
      const int ko = kk * 32;
      const v16h aq = load_a_frag(&Qs[tm * 16 * S] + ko, S, lane);
      const v16h ap = load_a_frag(&Ps[tm * 16 * S] + ko, S, lane);
      const v16h bp = load_b_frag(&Ps[tn * 16 * S] + ko, S, lane);
      const v16h bk = load_b_frag(&Ks[tn * 16 * S] + ko, S, lane);
      WMMA_F16(acc12, aq, bp);  // att1 = Q @ P^T
      WMMA_F16(acc12, ap, bk);  // att2 = P @ K^T
      WMMA_F16(accQK, aq, bk);  // raw Q @ K^T (cosine numerator)
      WMMA_F16(accPP, ap, bp);  // att_pos pre-Watt = P @ P^T
    }
    const int rbase = tm * 16 + ((lane >> 4) << 3);
    const int col = tn * 16 + (lane & 15);
    const float knc = kn[col];
#pragma unroll
    for (int e = 0; e < 8; e++) {
      const int row = rbase + e;
      const float g3 = accQK[e] * (qn[row] * knc);    // reciprocal-norm cosine
      Sf[row * SF + col] = (acc12[e] + g3) * 0.125f;  // / sqrt(DK)
      G4s[row * S + col] = (_Float16)accPP[e];
    }
  }
  __syncthreads();

  // ---------------- stage 2: att_pos = G4 @ Watt + batt; add box term ------
  const float scale = scaleS;
#pragma unroll
  for (int t = 0; t < 2; t++) {
    const int tile = wid * 2 + t;
    const int tm = tile >> 2, tn = tile & 3;
    v8f accAP = zero8();
#pragma unroll
    for (int kk = 0; kk < 2; kk++) {
      const int ko = kk * 32;
      const v16h a = load_a_frag(&G4s[tm * 16 * S] + ko, S, lane);
      const v16h w = load_b_frag(&WtT[tn * 16 * S] + ko, S, lane);
      WMMA_F16(accAP, a, w);
    }
    const int rbase = tm * 16 + ((lane >> 4) << 3);
    const int col = tn * 16 + (lane & 15);
    const float bt = battS[col];
#pragma unroll
    for (int e = 0; e < 8; e++) {
      const int row = rbase + e;
      float wg = 0.f;
      if (row < Nn && col < Nn) {
        const float x = boxg[(((size_t)(b * 16 + h)) * Nn + row) * Nn + col];
        wg = copysignf(log1pf(fabsf(x)), x);  // sign(x)*log1p(|x|)
      }
      Sf[row * SF + col] = (Sf[row * SF + col] + accAP[e] + bt + wg) * scale;
    }
  }
  __syncthreads();

  // ---------------- stage 3: row softmax (49 valid cols) -> f16 ------------
  if (tid < 64) {
    const int row = tid;
    if (row < Nn) {
      float m = -1e30f;
      for (int j = 0; j < Nn; j++) m = fmaxf(m, Sf[row * SF + j]);
      float s = 0.f;
      for (int j = 0; j < Nn; j++) s += __expf(Sf[row * SF + j] - m);
      const float inv = __builtin_amdgcn_rcpf(s);
      for (int j = 0; j < 64; j++) {
        const float p = (j < Nn) ? __expf(Sf[row * SF + j] - m) * inv : 0.f;
        S16[row * S + j] = (_Float16)p;
      }
    } else {
      for (int j = 0; j < 64; j++) S16[row * S + j] = (_Float16)0.f;
    }
  }
  __syncthreads();

  // ---------------- stage 4: out = softmax @ V (WMMA), store f16 -----------
#pragma unroll
  for (int t = 0; t < 2; t++) {
    const int tile = wid * 2 + t;
    const int tm = tile >> 2, tn = tile & 3;
    v8f accO = zero8();
#pragma unroll
    for (int kk = 0; kk < 2; kk++) {
      const int ko = kk * 32;
      const v16h a = load_a_frag(&S16[tm * 16 * S] + ko, S, lane);
      const v16h v = load_b_frag(&Vt[tn * 16 * S] + ko, S, lane);
      WMMA_F16(accO, a, v);
    }
    const int rbase = tm * 16 + ((lane >> 4) << 3);
    const int col = tn * 16 + (lane & 15);  // dv
#pragma unroll
    for (int e = 0; e < 8; e++) {
      const int row = rbase + e;
      if (row < Nn)
        AOg[gbase + (size_t)row * 1024 + col] = (_Float16)accO[e];
    }
  }
}

// ---------------------------------------------------------------------------
extern "C" void kernel_launch(void* const* d_in, const int* in_sizes, int n_in,
                              void* d_out, int out_size, void* d_ws, size_t ws_size,
                              hipStream_t stream) {
  (void)in_sizes; (void)n_in; (void)out_size; (void)ws_size;
  const float* queries = (const float*)d_in[0];
  const float* keys    = (const float*)d_in[1];
  const float* values  = (const float*)d_in[2];
  const float* gpos    = (const float*)d_in[3];
  const float* box     = (const float*)d_in[4];
  const float* Wq = (const float*)d_in[5];  const float* bq = (const float*)d_in[6];
  const float* Wk = (const float*)d_in[7];  const float* bk = (const float*)d_in[8];
  const float* Wv = (const float*)d_in[9];  const float* bv = (const float*)d_in[10];
  const float* Wo = (const float*)d_in[11]; const float* bo = (const float*)d_in[12];
  const float* Watt = (const float*)d_in[13];
  const float* batt = (const float*)d_in[14];
  const float* sp   = (const float*)d_in[15];

  const size_t MTOT = 512 * 49;  // 25088 (multiple of 128)
  const size_t DM = 1024;
  _Float16* Qh  = (_Float16*)d_ws;
  _Float16* Kh  = Qh + MTOT * DM;
  _Float16* Vh  = Kh + MTOT * DM;
  _Float16* AOh = Vh + MTOT * DM;   // total ws: 4 * 25088*1024 * 2B = ~196 MB

  const dim3 gg(1024 / 128, 25088 / 128);  // (8, 196)
  gemm_bias_kernel<float, _Float16><<<gg, 256, 0, stream>>>(queries, Wq, bq, Qh, 1024, 1024);
  gemm_bias_kernel<float, _Float16><<<gg, 256, 0, stream>>>(keys,    Wk, bk, Kh, 1024, 1024);
  gemm_bias_kernel<float, _Float16><<<gg, 256, 0, stream>>>(values,  Wv, bv, Vh, 1024, 1024);

  box_attention_kernel<<<dim3(16, 512), 256, 0, stream>>>(Qh, Kh, Vh, gpos, box,
                                                          Watt, batt, sp, AOh);

  gemm_bias_kernel<_Float16, float><<<gg, 256, 0, stream>>>(AOh, Wo, bo, (float*)d_out,
                                                            1024, 1024);
}
